// STGCNBlock_18734647345615
// MI455X (gfx1250) — compile-verified, matrix operand order checked by reading, not run on hardware
//
#include <hip/hip_runtime.h>
#include <hip/hip_bf16.h>

#define N_NODES 239616
#define N_EDGES 3833856
#define IN_C 64
#define OUT_C 128
#define L_LM 468
#define BN_EPS 1e-5f

typedef __bf16 v16bf __attribute__((ext_vector_type(16)));
typedef short  v16s  __attribute__((ext_vector_type(16)));
typedef float  v8f   __attribute__((ext_vector_type(8)));

union BF16x16 {
    v16bf bf;
    v16s  s;
    unsigned short u[16];
    uint4 q[2];
};

__device__ __forceinline__ unsigned short f2bf(float f) {
    unsigned int u = __float_as_uint(f);
    u += 0x7FFFu + ((u >> 16) & 1u);   // round-to-nearest-even
    return (unsigned short)(u >> 16);
}
__device__ __forceinline__ float bf2f(unsigned short h) {
    return __uint_as_float(((unsigned int)h) << 16);
}

__device__ __forceinline__ v8f wmma_bf16(v16bf a, v16bf b, v8f c) {
    return __builtin_amdgcn_wmma_f32_16x16x32_bf16(
        /*neg_a=*/false, a, /*neg_b=*/false, b,
        /*c_mod=*/(short)0, c, /*reuse_a=*/false, /*reuse_b=*/false);
}

// Map a row-major weight element (k, n) to its position in the swizzled
// B-fragment LDS image (fragments stored [frag][lane][16] so each lane's
// 16 bf16 values are 32 contiguous bytes -> 2x ds_load_b128).
// ISA 16-bit B 32x16 layout: lane%16 = column, lanes>=16 hold K 8..15 and 24..31.
__device__ __forceinline__ int swz_b(int k, int n, int kc_bits) {
    int nt = n >> 4, col = n & 15;
    int kc = k >> 5, kr = k & 31;
    int g = kr >> 3, sub = kr & 7;
    int lane = col + ((g & 1) << 4);
    int i = sub + ((g & 2) ? 8 : 0);
    return (((nt << kc_bits) | kc) * 32 + lane) * 16 + i;
}

// ---------------------------------------------------------------------------
// K0: zero agg accumulator and BN stats
// ---------------------------------------------------------------------------
__global__ void __launch_bounds__(256)
zero_kernel(float* __restrict__ agg, float* __restrict__ stats) {
    unsigned idx = blockIdx.x * 256u + threadIdx.x;
    if (idx < (unsigned)(N_NODES * IN_C)) agg[idx] = 0.0f;
    if (idx < 4 * OUT_C) stats[idx] = 0.0f;
}

// ---------------------------------------------------------------------------
// K1: edge scatter-add (segment_sum).  x and agg are L2-resident (61MB each).
// One thread = (edge, 4 channels): float4 gather + 4 global f32 atomics.
// ---------------------------------------------------------------------------
__global__ void __launch_bounds__(256)
scatter_kernel(const float* __restrict__ x, const long long* __restrict__ ei,
               float* __restrict__ agg) {
    size_t idx = (size_t)blockIdx.x * 256 + threadIdx.x;
    size_t e = idx >> 4;
    int c = (int)(idx & 15) << 2;
    long long src = ei[e];
    long long dst = ei[(size_t)N_EDGES + e];
    const float4 v = *reinterpret_cast<const float4*>(x + (size_t)src * IN_C + c);
    float* a = agg + (size_t)dst * IN_C + c;
    atomicAdd(a + 0, v.x);
    atomicAdd(a + 1, v.y);
    atomicAdd(a + 2, v.z);
    atomicAdd(a + 3, v.w);
}

// ---------------------------------------------------------------------------
// K2: h = agg@W_rel + x@W_root + b_rel  (bf16 out + BN1 stats)
//     r = x@res_w^T + res_b             (fp32, straight into d_out)
// block = 128 threads = 4 waves; each wave owns a 16x128 output tile.
// A fragments: 2x ds_load_b128 from row-major tiles (contiguous runs).
// B fragments: 2x ds_load_b128 from pre-swizzled weight image.
// Epilogue: single base pointer per wave, all stores immediate-offset.
// ---------------------------------------------------------------------------
__global__ void __launch_bounds__(128)
gemm_node_kernel(const float* __restrict__ x, const float* __restrict__ agg,
                 const float* __restrict__ W_rel, const float* __restrict__ b_rel,
                 const float* __restrict__ W_root,
                 const float* __restrict__ res_w, const float* __restrict__ res_b,
                 unsigned short* __restrict__ hbuf, float* __restrict__ rout,
                 float* __restrict__ stats1) {
    __shared__ __align__(16) unsigned short s_wA[IN_C * OUT_C];  // W_rel (then res_w^T), swizzled
    __shared__ __align__(16) unsigned short s_wB[IN_C * OUT_C];  // W_root, swizzled
    __shared__ __align__(16) unsigned short s_x[64 * IN_C];      // row-major
    __shared__ __align__(16) unsigned short s_agg[64 * IN_C];    // row-major
    __shared__ float s_sum[OUT_C];
    __shared__ float s_sq[OUT_C];

    const int tid = threadIdx.x;
    const unsigned rowbase = blockIdx.x * 64u;

    for (int i = tid; i < IN_C * OUT_C; i += 128) {
        int k = i >> 7, n = i & 127;
        int d = swz_b(k, n, 1);            // frag = nt*2 + kc   (K=64)
        s_wA[d] = f2bf(W_rel[i]);
        s_wB[d] = f2bf(W_root[i]);
    }
    for (int i = tid; i < 64 * IN_C; i += 128) {
        int r = i >> 6;
        unsigned g = (rowbase + r) * IN_C + (i & 63);
        s_x[i]   = f2bf(x[g]);
        s_agg[i] = f2bf(agg[g]);
    }
    if (tid < OUT_C) { s_sum[tid] = 0.0f; s_sq[tid] = 0.0f; }
    __syncthreads();

    const int lane = tid & 31, wid = tid >> 5;
    const int mloc = wid * 16 + (lane & 15);
    const int khi  = (lane & 16) ? 8 : 0;
    const int mbase = (lane & 16) ? 8 : 0;

    // A fragments: lane's 16 elems = two contiguous 16B runs (2x b128 each)
    BF16x16 ax[2], ag[2];
    for (int kc = 0; kc < 2; ++kc) {
        const uint4* px = reinterpret_cast<const uint4*>(&s_x[mloc * IN_C + kc * 32 + khi]);
        const uint4* pg = reinterpret_cast<const uint4*>(&s_agg[mloc * IN_C + kc * 32 + khi]);
        ax[kc].q[0] = px[0]; ax[kc].q[1] = px[2];
        ag[kc].q[0] = pg[0]; ag[kc].q[1] = pg[2];
    }

    // one base pointer per wave; every store below is base + constant offset
    const unsigned row0 = rowbase + wid * 16 + mbase;
    unsigned short* __restrict__ hp = hbuf + (size_t)row0 * OUT_C + (lane & 15);
    float* __restrict__ rp = rout + (size_t)row0 * OUT_C + (lane & 15);
    const float* __restrict__ brl = b_rel + (lane & 15);
    const float* __restrict__ brs = res_b + (lane & 15);
    float* __restrict__ ssum = s_sum + (lane & 15);
    float* __restrict__ ssq  = s_sq + (lane & 15);

#pragma unroll
    for (int nt = 0; nt < 8; ++nt) {
        v8f acc = {};
#pragma unroll
        for (int kc = 0; kc < 2; ++kc) {
            BF16x16 b0, b1;
            const uint4* p0 = reinterpret_cast<const uint4*>(&s_wA[((nt * 2 + kc) * 32 + lane) * 16]);
            const uint4* p1 = reinterpret_cast<const uint4*>(&s_wB[((nt * 2 + kc) * 32 + lane) * 16]);
            b0.q[0] = p0[0]; b0.q[1] = p0[1];
            b1.q[0] = p1[0]; b1.q[1] = p1[1];
            acc = wmma_bf16(ag[kc].bf, b0.bf, acc);
            acc = wmma_bf16(ax[kc].bf, b1.bf, acc);
        }
        float bias = brl[nt * 16];
        float lsum = 0.0f, lsq = 0.0f;
#pragma unroll
        for (int j = 0; j < 8; ++j) {
            float v = acc[j] + bias;
            lsum += v; lsq += v * v;
            hp[nt * 16 + j * OUT_C] = f2bf(v);
        }
        atomicAdd(ssum + nt * 16, lsum);
        atomicAdd(ssq + nt * 16, lsq);
    }

    __syncthreads();
    // restage s_wA with residual weight transposed to [ci][co], swizzled
    for (int i = tid; i < IN_C * OUT_C; i += 128) {
        int k = i >> 7, n = i & 127;
        s_wA[swz_b(k, n, 1)] = f2bf(res_w[n * IN_C + k]);
    }
    __syncthreads();

#pragma unroll
    for (int nt = 0; nt < 8; ++nt) {
        v8f acc = {};
#pragma unroll
        for (int kc = 0; kc < 2; ++kc) {
            BF16x16 b0;
            const uint4* p0 = reinterpret_cast<const uint4*>(&s_wA[((nt * 2 + kc) * 32 + lane) * 16]);
            b0.q[0] = p0[0]; b0.q[1] = p0[1];
            acc = wmma_bf16(ax[kc].bf, b0.bf, acc);
        }
        float bias = brs[nt * 16];
#pragma unroll
        for (int j = 0; j < 8; ++j)
            rp[nt * 16 + j * OUT_C] = acc[j] + bias;
    }

    __syncthreads();
    if (tid < OUT_C) {
        atomicAdd(&stats1[tid], s_sum[tid]);
        atomicAdd(&stats1[OUT_C + tid], s_sq[tid]);
    }
}

// ---------------------------------------------------------------------------
// K3: y = Conv1d(relu(BN1(h)), k=3, pad=1) + conv_b   (bf16 out + BN2 stats)
// Conv as 3 shifted GEMMs (K=128, 4 chunks of 32).  Activation rows + halo
// staged row-major in LDS (A frags = 2x b128); per-tap weights staged
// swizzled (B frags = 2x b128).  Boundary taps zeroed per-lane (EXEC
// uniform for WMMA).
// ---------------------------------------------------------------------------
__global__ void __launch_bounds__(128)
conv_tcn_kernel(const unsigned short* __restrict__ hbuf,
                const float* __restrict__ conv_w, const float* __restrict__ conv_b,
                const float* __restrict__ bn1_g, const float* __restrict__ bn1_b,
                const float* __restrict__ stats1,
                unsigned short* __restrict__ ybuf, float* __restrict__ stats2) {
    __shared__ __align__(16) unsigned short s_act[66 * OUT_C];   // rows rowbase-1..rowbase+64
    __shared__ __align__(16) unsigned short s_w[OUT_C * OUT_C];  // one tap, swizzled
    __shared__ float s_scale[OUT_C], s_shift[OUT_C], s_sum[OUT_C], s_sq[OUT_C];

    const int tid = threadIdx.x;
    const unsigned rowbase = blockIdx.x * 64u;
    const float invN = 1.0f / (float)N_NODES;

    if (tid < OUT_C) {
        float mean = stats1[tid] * invN;
        float var  = stats1[OUT_C + tid] * invN - mean * mean;
        float sc = bn1_g[tid] * rsqrtf(var + BN_EPS);
        s_scale[tid] = sc;
        s_shift[tid] = bn1_b[tid] - mean * sc;
        s_sum[tid] = 0.0f; s_sq[tid] = 0.0f;
    }
    __syncthreads();

    for (int i = tid; i < 66 * OUT_C; i += 128) {
        int rl = i >> 7, c = i & 127;
        int g = (int)rowbase - 1 + rl;
        float v = 0.0f;
        if (g >= 0 && g < N_NODES) {
            float f = bf2f(hbuf[(unsigned)g * OUT_C + c]);
            v = f * s_scale[c] + s_shift[c];
            v = v > 0.0f ? v : 0.0f;
        }
        s_act[i] = f2bf(v);
    }

    const int lane = tid & 31, wid = tid >> 5;
    const int m = wid * 16 + (lane & 15);
    const int l = (int)((rowbase + m) % L_LM);
    const int mbase = (lane & 16) ? 8 : 0;
    const int khi  = (lane & 16) ? 8 : 0;
    const uint4 z4 = {0u, 0u, 0u, 0u};

    v8f acc[8] = {};

    for (int k = 0; k < 3; ++k) {
        __syncthreads();   // act staged / previous tap weights consumed
        for (int i = tid; i < OUT_C * OUT_C; i += 128) {
            int ci = i >> 7, co = i & 127;
            s_w[swz_b(ci, co, 2)] = f2bf(conv_w[co * (OUT_C * 3) + ci * 3 + k]);
        }
        __syncthreads();

        bool valid = !((l == 0 && k == 0) || (l == L_LM - 1 && k == 2));
        BF16x16 a[4];
#pragma unroll
        for (int kc = 0; kc < 4; ++kc) {
            const uint4* pa = reinterpret_cast<const uint4*>(
                &s_act[(m + k) * OUT_C + kc * 32 + khi]);
            a[kc].q[0] = pa[0];
            a[kc].q[1] = pa[2];
            if (!valid) { a[kc].q[0] = z4; a[kc].q[1] = z4; }
        }
#pragma unroll
        for (int nt = 0; nt < 8; ++nt) {
#pragma unroll
            for (int kc = 0; kc < 4; ++kc) {
                BF16x16 b;
                const uint4* pb = reinterpret_cast<const uint4*>(
                    &s_w[((nt * 4 + kc) * 32 + lane) * 16]);
                b.q[0] = pb[0]; b.q[1] = pb[1];
                acc[nt] = wmma_bf16(a[kc].bf, b.bf, acc[nt]);
            }
        }
    }

    const unsigned row0 = rowbase + wid * 16 + mbase;
    unsigned short* __restrict__ yp = ybuf + (size_t)row0 * OUT_C + (lane & 15);
    const float* __restrict__ cb = conv_b + (lane & 15);
    float* __restrict__ ssum = s_sum + (lane & 15);
    float* __restrict__ ssq  = s_sq + (lane & 15);

#pragma unroll
    for (int nt = 0; nt < 8; ++nt) {
        float bias = cb[nt * 16];
        float lsum = 0.0f, lsq = 0.0f;
#pragma unroll
        for (int j = 0; j < 8; ++j) {
            float v = acc[nt][j] + bias;
            lsum += v; lsq += v * v;
            yp[nt * 16 + j * OUT_C] = f2bf(v);
        }
        atomicAdd(ssum + nt * 16, lsum);
        atomicAdd(ssq + nt * 16, lsq);
    }
    __syncthreads();
    if (tid < OUT_C) {
        atomicAdd(&stats2[tid], s_sum[tid]);
        atomicAdd(&stats2[OUT_C + tid], s_sq[tid]);
    }
}

// ---------------------------------------------------------------------------
// K4: out = relu(BN2(y)) + r   (r already resides in d_out)
// Channel index is loop-invariant per thread -> scale/shift hoisted.
// ---------------------------------------------------------------------------
__global__ void __launch_bounds__(256)
bn2_residual_kernel(const unsigned short* __restrict__ ybuf,
                    const float* __restrict__ stats2,
                    const float* __restrict__ bn2_g, const float* __restrict__ bn2_b,
                    float* __restrict__ out) {
    __shared__ float s_scale[OUT_C], s_shift[OUT_C];
    const int tid = threadIdx.x;
    if (tid < OUT_C) {
        const float invN = 1.0f / (float)N_NODES;
        float mean = stats2[tid] * invN;
        float var  = stats2[OUT_C + tid] * invN - mean * mean;
        float sc = bn2_g[tid] * rsqrtf(var + BN_EPS);
        s_scale[tid] = sc;
        s_shift[tid] = bn2_b[tid] - mean * sc;
    }
    __syncthreads();
    const unsigned base = blockIdx.x * 2048u + tid;
    const int c = (int)(base & 127);          // invariant: stride 256 keeps channel
    const float sc = s_scale[c], sh = s_shift[c];
    const unsigned short* __restrict__ yp = ybuf + base;
    float* __restrict__ op = out + base;
#pragma unroll
    for (int it = 0; it < 8; ++it) {
        float v = bf2f(yp[it * 256]) * sc + sh;
        v = v > 0.0f ? v : 0.0f;
        op[it * 256] += v;
    }
}

// ---------------------------------------------------------------------------
extern "C" void kernel_launch(void* const* d_in, const int* in_sizes, int n_in,
                              void* d_out, int out_size, void* d_ws, size_t ws_size,
                              hipStream_t stream) {
    (void)in_sizes; (void)n_in; (void)out_size; (void)ws_size;
    const float*     x      = (const float*)d_in[0];
    const long long* ei     = (const long long*)d_in[1];   // int64 edge_index
    const float*     W_rel  = (const float*)d_in[3];
    const float*     b_rel  = (const float*)d_in[4];
    const float*     W_root = (const float*)d_in[5];
    const float*     bn1_g  = (const float*)d_in[6];
    const float*     bn1_b  = (const float*)d_in[7];
    const float*     conv_w = (const float*)d_in[8];
    const float*     conv_b = (const float*)d_in[9];
    const float*     bn2_g  = (const float*)d_in[10];
    const float*     bn2_b  = (const float*)d_in[11];
    const float*     res_w  = (const float*)d_in[12];
    const float*     res_b  = (const float*)d_in[13];
    float* out = (float*)d_out;

    char* ws = (char*)d_ws;
    const size_t aggBytes = (size_t)N_NODES * IN_C * sizeof(float);           // 61.3 MB
    const size_t hBytes   = (size_t)N_NODES * OUT_C * sizeof(unsigned short); // 61.3 MB
    float*          agg    = (float*)ws;
    unsigned short* hbuf   = (unsigned short*)(ws + aggBytes);
    unsigned short* ybuf   = (unsigned short*)(ws + aggBytes + hBytes);
    float*          stats  = (float*)(ws + aggBytes + 2 * hBytes);
    float*          stats1 = stats;             // [sum1(128), sq1(128)]
    float*          stats2 = stats + 2 * OUT_C; // [sum2(128), sq2(128)]

    zero_kernel<<<(unsigned)(((size_t)N_NODES * IN_C + 255) / 256), 256, 0, stream>>>(agg, stats);
    scatter_kernel<<<(unsigned)((size_t)N_EDGES * 16 / 256), 256, 0, stream>>>(x, ei, agg);
    gemm_node_kernel<<<N_NODES / 64, 128, 0, stream>>>(
        x, agg, W_rel, b_rel, W_root, res_w, res_b, hbuf, out, stats1);
    conv_tcn_kernel<<<N_NODES / 64, 128, 0, stream>>>(
        hbuf, conv_w, conv_b, bn1_g, bn1_b, stats1, ybuf, stats2);
    bn2_residual_kernel<<<(unsigned)((size_t)N_NODES * OUT_C / 2048), 256, 0, stream>>>(
        ybuf, stats2, bn2_g, bn2_b, out);
}
